// RNN_VAE_90452011254343
// MI455X (gfx1250) — matrix operand order repaired and checked by reading, not compiled, not run on hardware
//
#include <hip/hip_runtime.h>
#include <hip/hip_bf16.h>

#define B_     1024
#define S_     512
#define INPUT_ 40
#define HID_   512
#define LAT_   128

typedef __attribute__((ext_vector_type(16))) __bf16 v16b;
typedef __attribute__((ext_vector_type(8)))  float  v8f;

// ---------------------------------------------------------------------------
// helpers
// ---------------------------------------------------------------------------
__device__ __forceinline__ float sigm_f(float x) {
    return 1.0f / (1.0f + __expf(-x));
}
__device__ __forceinline__ float tanh_f(float x) {
    x = fminf(15.0f, fmaxf(-15.0f, x));
    float e = __expf(2.0f * x);
    return (e - 1.0f) / (e + 1.0f);
}

__device__ __forceinline__ v8f wmma_bf16(v16b a, v16b b, v8f c) {
    return __builtin_amdgcn_wmma_f32_16x16x32_bf16(
        /*neg_a=*/false, a, /*neg_b=*/false, b,
        /*c_mod=*/(short)0, c, /*reuse_a=*/false, /*reuse_b=*/false);
}

// A fragment: 16x32 (MxK) bf16, row-major source with given stride.
// lane L: row = L&15, kh = L>>4. elems 0..7 -> K = kh*8 + e,
// elems 8..15 -> K = 16 + kh*8 + (e-8).   (ISA 7.12.2, 16-bit A 16x32)
__device__ __forceinline__ v16b load_a_frag(const __bf16* base, int stride, int lane) {
    int row = lane & 15;
    int kh  = lane >> 4;
    const __bf16* p = base + row * stride + kh * 8;
    union { v16b v; uint4 q[2]; } u;
    u.q[0] = *(const uint4*)(p);       // K = kh*8 .. +7
    u.q[1] = *(const uint4*)(p + 16);  // K = 16+kh*8 .. +7
    return u.v;
}

// B fragment: 32x16 (KxN) bf16 where B = W^T and W rows are contiguous in K.
// lane L: col = L&15, kh = L>>4. elems 0..7 -> K = kh*8 + e,
// elems 8..15 -> K = 16 + kh*8 + (e-8).
__device__ __forceinline__ v16b load_b_frag(const __bf16* wrow, int k0, int lane) {
    const __bf16* p = wrow + k0 + (lane >> 4) * 8;
    union { v16b v; uint4 q[2]; } u;
    u.q[0] = *(const uint4*)(p);
    u.q[1] = *(const uint4*)(p + 16);
    return u.v;
}

// One A-segment of a 4-gate GEMM, software-pipelined 1 k-tile deep so the
// next tile's A/B loads overlap the current tile's 4 WMMAs.
__device__ __forceinline__ void gemm_seg_pipe(
    const __bf16* a_base, int stride, int nt,
    const __bf16* w0, const __bf16* w1, const __bf16* w2, const __bf16* w3,
    int& k0, int lane, v8f acc[4])
{
    v16b a  = load_a_frag(a_base, stride, lane);
    v16b b0 = load_b_frag(w0, k0, lane);
    v16b b1 = load_b_frag(w1, k0, lane);
    v16b b2 = load_b_frag(w2, k0, lane);
    v16b b3 = load_b_frag(w3, k0, lane);
#pragma unroll
    for (int kt = 1; kt < nt; ++kt) {
        int k1 = k0 + kt * 32;
        v16b an  = load_a_frag(a_base + kt * 32, stride, lane);
        v16b bn0 = load_b_frag(w0, k1, lane);
        v16b bn1 = load_b_frag(w1, k1, lane);
        v16b bn2 = load_b_frag(w2, k1, lane);
        v16b bn3 = load_b_frag(w3, k1, lane);
        acc[0] = wmma_bf16(a, b0, acc[0]);
        acc[1] = wmma_bf16(a, b1, acc[1]);
        acc[2] = wmma_bf16(a, b2, acc[2]);
        acc[3] = wmma_bf16(a, b3, acc[3]);
        a = an; b0 = bn0; b1 = bn1; b2 = bn2; b3 = bn3;
    }
    acc[0] = wmma_bf16(a, b0, acc[0]);
    acc[1] = wmma_bf16(a, b1, acc[1]);
    acc[2] = wmma_bf16(a, b2, acc[2]);
    acc[3] = wmma_bf16(a, b3, acc[3]);
    k0 += nt * 32;
}

// Full 4-gate GEMM for one 16-wide hidden chunk over two A segments.
// W packed gate-major: row (g*Hpad + col), K contiguous with stride Kstr.
__device__ __forceinline__ void gemm4(
    const __bf16* a1, int st1, int nt1,
    const __bf16* a2, int st2, int nt2,
    const __bf16* W, int Kstr, int Hpad,
    int ht, int lane, v8f acc[4])
{
    int col = ht * 16 + (lane & 15);
    const __bf16* w0 = W + (size_t)(0 * Hpad + col) * Kstr;
    const __bf16* w1 = W + (size_t)(1 * Hpad + col) * Kstr;
    const __bf16* w2 = W + (size_t)(2 * Hpad + col) * Kstr;
    const __bf16* w3 = W + (size_t)(3 * Hpad + col) * Kstr;
    int k0 = 0;
    gemm_seg_pipe(a1, st1, nt1, w0, w1, w2, w3, k0, lane, acc);
    gemm_seg_pipe(a2, st2, nt2, w0, w1, w2, w3, k0, lane, acc);
}

// LSTM pointwise tail in C/D register layout (row = 8*(lane>>4)+r)
__device__ __forceinline__ void cell_tail(
    v8f acc[4], const float* bias, int Hpad, int ht, int lane,
    v8f& c, float hout[8])
{
    int col = ht * 16 + (lane & 15);
    float bi = bias[0 * Hpad + col];
    float bf = bias[1 * Hpad + col];
    float bg = bias[2 * Hpad + col];
    float bo = bias[3 * Hpad + col];
#pragma unroll
    for (int r = 0; r < 8; ++r) {
        float iv = sigm_f(acc[0][r] + bi);
        float fv = sigm_f(acc[1][r] + bf);
        float gv = tanh_f(acc[2][r] + bg);
        float ov = sigm_f(acc[3][r] + bo);
        float cn = fv * c[r] + iv * gv;
        c[r] = cn;
        hout[r] = ov * tanh_f(cn);
    }
}

__device__ __forceinline__ void store_h(
    __bf16* dst, int stride, int ht, int lane, const float hv[8])
{
    int col = ht * 16 + (lane & 15);
    int rb  = (lane >> 4) * 8;
#pragma unroll
    for (int r = 0; r < 8; ++r)
        dst[(rb + r) * stride + col] = (__bf16)hv[r];
}

// ---------------------------------------------------------------------------
// weight / bias prepack kernels
// ---------------------------------------------------------------------------
__global__ void pack_w_kernel(const float* __restrict__ W1,
                              const float* __restrict__ W2,
                              __bf16* __restrict__ dst,
                              int Hpad, int Hvalid,
                              int K1, int K1pad, int K2, int Kp)
{
    long total = (long)(4 * Hpad) * Kp;
    for (long idx = (long)blockIdx.x * blockDim.x + threadIdx.x; idx < total;
         idx += (long)gridDim.x * blockDim.x) {
        int n = (int)(idx / Kp);
        int k = (int)(idx % Kp);
        int g = n / Hpad, rr = n % Hpad;
        float v = 0.0f;
        if (rr < Hvalid) {
            int orig = g * Hvalid + rr;
            if (k < K1)                            v = W1[(size_t)orig * K1 + k];
            else if (k >= K1pad && k < K1pad + K2) v = W2[(size_t)orig * K2 + (k - K1pad)];
        }
        dst[idx] = (__bf16)v;
    }
}

__global__ void pack_b_kernel(const float* __restrict__ b1,
                              const float* __restrict__ b2,
                              float* __restrict__ dst, int Hpad, int Hvalid)
{
    int n = blockIdx.x * blockDim.x + threadIdx.x;
    if (n >= 4 * Hpad) return;
    int g = n / Hpad, rr = n % Hpad;
    dst[n] = (rr < Hvalid) ? (b1[g * Hvalid + rr] + b2[g * Hvalid + rr]) : 0.0f;
}

// ---------------------------------------------------------------------------
// persistent VAE-LSTM kernel: one block = 16 batch rows, 16 waves, full seq.
// Column-split: enc_h/d1 -> 2 hidden chunks per wave; mu on waves 0-7 runs
// concurrently with logvar on waves 8-15; d2 on waves 13-15.
// ---------------------------------------------------------------------------
__global__ __launch_bounds__(512, 1)
void vae_lstm_kernel(const float* __restrict__ x,
                     const float* __restrict__ eps,
                     const __bf16* __restrict__ Wh,  const float* __restrict__ bh,
                     const __bf16* __restrict__ Wmu, const float* __restrict__ bmu,
                     const __bf16* __restrict__ Wlv, const float* __restrict__ blv,
                     const __bf16* __restrict__ Wd1, const float* __restrict__ bd1,
                     const __bf16* __restrict__ Wd2, const float* __restrict__ bd2,
                     float* __restrict__ recon,
                     float* __restrict__ mu_out,
                     float* __restrict__ lv_out)
{
    __shared__ __align__(16) __bf16 s_x[16 * 64];          // x_t, K-padded 40->64
    __shared__ __align__(16) __bf16 s_hh [2][16 * 512];
    __shared__ __align__(16) __bf16 s_hmu[2][16 * 128];
    __shared__ __align__(16) __bf16 s_hlv[2][16 * 128];
    __shared__ __align__(16) __bf16 s_z  [16 * 128];
    __shared__ __align__(16) __bf16 s_hd1[2][16 * 512];
    __shared__ __align__(16) __bf16 s_hd2[2][16 * 64];     // 40->48 valid, 64 stride
    __shared__ __align__(16) float  s_lvf[16 * 128];       // f32 logvar handoff

    const int tid  = threadIdx.x;
    const int lane = tid & 31;
    const int w    = tid >> 5;          // wave id 0..15
    const int blk  = blockIdx.x;        // 16-row batch tile

    // zero-init shared state (t=0 states zero; also K/row padding)
    for (int i = tid; i < 16 * 64; i += 512) { s_x[i] = (__bf16)0.0f;
                                               s_hd2[0][i] = (__bf16)0.0f; s_hd2[1][i] = (__bf16)0.0f; }
    for (int i = tid; i < 16 * 512; i += 512) { s_hh[0][i] = (__bf16)0.0f; s_hh[1][i] = (__bf16)0.0f;
                                                s_hd1[0][i] = (__bf16)0.0f; s_hd1[1][i] = (__bf16)0.0f; }
    for (int i = tid; i < 16 * 128; i += 512) { s_hmu[0][i] = (__bf16)0.0f; s_hmu[1][i] = (__bf16)0.0f;
                                                s_hlv[0][i] = (__bf16)0.0f; s_hlv[1][i] = (__bf16)0.0f;
                                                s_z[i] = (__bf16)0.0f; }
    __syncthreads();

    // persistent cell state (f32, WMMA C/D layout) — registers only
    v8f c_h[2] = {}; v8f c_d1[2] = {};
    v8f c_ml = {};   // c_mu (waves 0-7) or c_lv (waves 8-15)
    v8f c_d2 = {};   // waves 13-15

    const int rb = (lane >> 4) * 8;     // C/D row base for this lane

#pragma unroll 1
    for (int t = 0; t < S_; ++t) {
        const int cur = t & 1, prv = cur ^ 1;

        // ---- stage 1: stage x_t into LDS (bf16, K-padded) ----
        for (int i = tid; i < 16 * INPUT_; i += 512) {
            int row = i / INPUT_, k = i % INPUT_;
            int b = blk * 16 + row;
            s_x[row * 64 + k] = (__bf16)x[((size_t)b * S_ + t) * INPUT_ + k];
        }
        __syncthreads();

        // ---- stage 2: encoder hidden cell (40->512), K = 64 + 512 ----
#pragma unroll 1
        for (int cc = 0; cc < 2; ++cc) {
            int ht = w + 16 * cc;                      // hidden tiles 0..31
            v8f acc[4] = {};
            gemm4(s_x, 64, 2, s_hh[prv], 512, 16, Wh, 576, 512, ht, lane, acc);
            float hv[8];
            cell_tail(acc, bh, 512, ht, lane, c_h[cc], hv);
            store_h(s_hh[cur], 512, ht, lane, hv);
        }
        __syncthreads();

        // ---- stage 3: mu (waves 0-7) || logvar (waves 8-15), K = 512+128 ----
        float hml[8];
        if (w < 8) {
            v8f acc[4] = {};
            gemm4(s_hh[cur], 512, 16, s_hmu[prv], 128, 4, Wmu, 640, 128, w, lane, acc);
            cell_tail(acc, bmu, 128, w, lane, c_ml, hml);
            store_h(s_hmu[cur], 128, w, lane, hml);
        } else {
            int ht = w - 8;
            v8f acc[4] = {};
            gemm4(s_hh[cur], 512, 16, s_hlv[prv], 128, 4, Wlv, 640, 128, ht, lane, acc);
            cell_tail(acc, blv, 128, ht, lane, c_ml, hml);
            store_h(s_hlv[cur], 128, ht, lane, hml);
            int col = ht * 16 + (lane & 15);
#pragma unroll
            for (int r = 0; r < 8; ++r) {
                int row = rb + r;
                int b = blk * 16 + row;
                s_lvf[row * 128 + col] = hml[r];                      // f32 handoff
                lv_out[((size_t)b * S_ + t) * LAT_ + col] = hml[r];
            }
        }
        __syncthreads();

        // z = mu + eps * exp(0.5*logvar)   (mu waves; full f32 precision)
        if (w < 8) {
            int col = w * 16 + (lane & 15);
#pragma unroll
            for (int r = 0; r < 8; ++r) {
                int row = rb + r;
                int b = blk * 16 + row;
                size_t gi = ((size_t)b * S_ + t) * LAT_ + col;
                mu_out[gi] = hml[r];
                float zv = hml[r] + eps[gi] * __expf(0.5f * s_lvf[row * 128 + col]);
                s_z[row * 128 + col] = (__bf16)zv;
            }
        }
        __syncthreads();

        // ---- stage 4: decoder cell 1 (128->512), K = 128 + 512 ----
#pragma unroll 1
        for (int cc = 0; cc < 2; ++cc) {
            int ht = w + 16 * cc;
            v8f acc[4] = {};
            gemm4(s_z, 128, 4, s_hd1[prv], 512, 16, Wd1, 640, 512, ht, lane, acc);
            float hv[8];
            cell_tail(acc, bd1, 512, ht, lane, c_d1[cc], hv);
            store_h(s_hd1[cur], 512, ht, lane, hv);
        }
        __syncthreads();

        // ---- stage 5: decoder cell 2 (512->40, padded 48), K = 512 + 64 ----
        if (w >= 13) {
            int ht = w - 13;                           // hidden tiles 0..2
            v8f acc[4] = {};
            gemm4(s_hd1[cur], 512, 16, s_hd2[prv], 64, 2, Wd2, 576, 48, ht, lane, acc);
            float hv[8];
            cell_tail(acc, bd2, 48, ht, lane, c_d2, hv);
            store_h(s_hd2[cur], 64, ht, lane, hv);
            int col = ht * 16 + (lane & 15);
            if (col < INPUT_) {
#pragma unroll
                for (int r = 0; r < 8; ++r) {
                    int b = blk * 16 + rb + r;
                    recon[((size_t)b * S_ + t) * INPUT_ + col] = hv[r];
                }
            }
        }
        __syncthreads();
    }
}

// ---------------------------------------------------------------------------
// launch
// ---------------------------------------------------------------------------
extern "C" void kernel_launch(void* const* d_in, const int* in_sizes, int n_in,
                              void* d_out, int out_size, void* d_ws, size_t ws_size,
                              hipStream_t stream) {
    (void)in_sizes; (void)n_in; (void)out_size; (void)ws_size;
    const float* x      = (const float*)d_in[0];
    const float* eps    = (const float*)d_in[1];
    const float* Wih_h  = (const float*)d_in[2];
    const float* Whh_h  = (const float*)d_in[3];
    const float* bih_h  = (const float*)d_in[4];
    const float* bhh_h  = (const float*)d_in[5];
    const float* Wih_mu = (const float*)d_in[6];
    const float* Whh_mu = (const float*)d_in[7];
    const float* bih_mu = (const float*)d_in[8];
    const float* bhh_mu = (const float*)d_in[9];
    const float* Wih_lv = (const float*)d_in[10];
    const float* Whh_lv = (const float*)d_in[11];
    const float* bih_lv = (const float*)d_in[12];
    const float* bhh_lv = (const float*)d_in[13];
    const float* Wih_d1 = (const float*)d_in[14];
    const float* Whh_d1 = (const float*)d_in[15];
    const float* bih_d1 = (const float*)d_in[16];
    const float* bhh_d1 = (const float*)d_in[17];
    const float* Wih_d2 = (const float*)d_in[18];
    const float* Whh_d2 = (const float*)d_in[19];
    const float* bih_d2 = (const float*)d_in[20];
    const float* bhh_d2 = (const float*)d_in[21];

    // workspace layout (bytes, 256-aligned)
    char* ws = (char*)d_ws;
    __bf16* Wh  = (__bf16*)(ws + 0);          // 2048 x 576
    __bf16* Wmu = (__bf16*)(ws + 2359296);    //  512 x 640
    __bf16* Wlv = (__bf16*)(ws + 3014656);    //  512 x 640
    __bf16* Wd1 = (__bf16*)(ws + 3670016);    // 2048 x 640
    __bf16* Wd2 = (__bf16*)(ws + 6291456);    //  192 x 576
    float*  bh  = (float*) (ws + 6512640);    // 2048
    float*  bmu = (float*) (ws + 6520832);    //  512
    float*  blv = (float*) (ws + 6522880);    //  512
    float*  bd1 = (float*) (ws + 6524928);    // 2048
    float*  bd2 = (float*) (ws + 6533120);    //  192

    auto gw = [](long total) { return (int)((total + 255) / 256); };
    pack_w_kernel<<<gw(2048L * 576), 256, 0, stream>>>(Wih_h,  Whh_h,  Wh,  512, 512, 40,  64,  512, 576);
    pack_w_kernel<<<gw(512L  * 640), 256, 0, stream>>>(Wih_mu, Whh_mu, Wmu, 128, 128, 512, 512, 128, 640);
    pack_w_kernel<<<gw(512L  * 640), 256, 0, stream>>>(Wih_lv, Whh_lv, Wlv, 128, 128, 512, 512, 128, 640);
    pack_w_kernel<<<gw(2048L * 640), 256, 0, stream>>>(Wih_d1, Whh_d1, Wd1, 512, 512, 128, 128, 512, 640);
    pack_w_kernel<<<gw(192L  * 576), 256, 0, stream>>>(Wih_d2, Whh_d2, Wd2, 48,  40,  512, 512, 40,  576);
    pack_b_kernel<<<8, 256, 0, stream>>>(bih_h,  bhh_h,  bh,  512, 512);
    pack_b_kernel<<<2, 256, 0, stream>>>(bih_mu, bhh_mu, bmu, 128, 128);
    pack_b_kernel<<<2, 256, 0, stream>>>(bih_lv, bhh_lv, blv, 128, 128);
    pack_b_kernel<<<8, 256, 0, stream>>>(bih_d1, bhh_d1, bd1, 512, 512);
    pack_b_kernel<<<1, 256, 0, stream>>>(bih_d2, bhh_d2, bd2, 48,  40);

    // outputs: recon [B,S,40] | mu [B,S,128] | logvar [B,S,128], f32 flat
    float* recon = (float*)d_out;
    float* mu_o  = recon + (size_t)B_ * S_ * INPUT_;
    float* lv_o  = mu_o  + (size_t)B_ * S_ * LAT_;

    vae_lstm_kernel<<<B_ / 16, 512, 0, stream>>>(
        x, eps, Wh, bh, Wmu, bmu, Wlv, blv, Wd1, bd1, Wd2, bd2,
        recon, mu_o, lv_o);
}
